// Attention_55490977465075
// MI455X (gfx1250) — compile-verified, matrix operand order checked by reading, not run on hardware
//
#include <hip/hip_runtime.h>
#include <hip/hip_bf16.h>

typedef __attribute__((ext_vector_type(16))) _Float16 v16h;
typedef __attribute__((ext_vector_type(8)))  _Float16 v8h;
typedef __attribute__((ext_vector_type(4)))  _Float16 v4h;
typedef __attribute__((ext_vector_type(8)))  float    v8f;
typedef __attribute__((ext_vector_type(4)))  int      v4i;

#define WMMA_F32_F16(a, b, c) \
  __builtin_amdgcn_wmma_f32_16x16x32_f16(false, (a), false, (b), (short)0, (c), false, false)

// gfx1250 async global->LDS path (guarded: degrades to plain b128 copies)
#if defined(__gfx1250__) && \
    __has_builtin(__builtin_amdgcn_global_load_async_to_lds_b128) && \
    __has_builtin(__builtin_amdgcn_s_wait_asynccnt)
#define USE_ASYNC 1
#else
#define USE_ASYNC 0
#endif

__device__ __forceinline__ void copy_b128(const _Float16* g, _Float16* l) {
#if USE_ASYNC
  __builtin_amdgcn_global_load_async_to_lds_b128(
      (__attribute__((address_space(1))) v4i*)(void*)g,
      (__attribute__((address_space(3))) v4i*)(void*)l, 0, 0);
#else
  *(v8h*)l = *(const v8h*)g;
#endif
}

__device__ __forceinline__ void wait_async() {
#if USE_ASYNC
  __builtin_amdgcn_s_wait_asynccnt(0);
#endif
}

static constexpr int BSZ = 2, T = 2048, CD = 1024, H = 16, D = 64;
static constexpr int MROWS = BSZ * T;                 // 4096
static constexpr long QKVSZ = (long)BSZ * H * T * D;  // 4,194,304 f16 elems

// Per ISA 16-bit A/B fragment layout: halves 0..7 hold K = 8*g..8*g+7,
// halves 8..15 hold K = 16+8*g..23+8*g (g = lane>>4); per-lane row = lane&15.
// Works for any fragment whose per-lane row is contiguous in memory.
__device__ __forceinline__ v16h load_frag(const _Float16* row, int g) {
  v8h lo = *(const v8h*)(row + 8 * g);
  v8h hi = *(const v8h*)(row + 16 + 8 * g);
  v16h r;
#pragma unroll
  for (int i = 0; i < 8; ++i) { r[i] = lo[i]; r[i + 8] = hi[i]; }
  return r;
}

// ---------------------------------------------------------------------------
// fp32 -> f16 conversion pre-pass (vectorized x4)
// ---------------------------------------------------------------------------
__global__ __launch_bounds__(256) void cvt_f16_kernel(
    const float* __restrict__ s, _Float16* __restrict__ d, int n4) {
  int i = blockIdx.x * 256 + threadIdx.x;
  if (i < n4) {
    float4 f = ((const float4*)s)[i];
    v4h o = {(_Float16)f.x, (_Float16)f.y, (_Float16)f.z, (_Float16)f.w};
    ((v4h*)d)[i] = o;
  }
}

// ---------------------------------------------------------------------------
// GEMM: Y[4096 x N] = A[4096 x K] * W[K x N] + bias   (A, W are f16)
// Block: 256 threads (8 waves), tile 128(M) x 128(N), K-step 32,
// double-buffered LDS, wave tile 64x32 -> 8 WMMA per K-step.
// TO_QKV=true : scatter f16 q/k/v into [B,H,T,D] ws.  else fp32 row-major out.
// ---------------------------------------------------------------------------
template <int K, int N, bool TO_QKV>
__global__ __launch_bounds__(256) void gemm_wmma_kernel(
    const _Float16* __restrict__ A, const _Float16* __restrict__ W,
    const float* __restrict__ bias, void* __restrict__ outp) {
  __shared__ __align__(16) _Float16 Alds[2][128][40];  // [m][k], 80B stride
  __shared__ __align__(16) _Float16 Blds[2][128][40];  // [n][k]

  const int tid  = threadIdx.x;
  const int lane = tid & 31;
  const int ln   = lane & 15;
  const int g    = lane >> 4;
  const int wave = tid >> 5;   // 0..7
  const int wm   = wave >> 2;  // 0..1 : 64-row strip
  const int wn   = wave & 3;   // 0..3 : 32-col strip

  const int m0 = blockIdx.y * 128;
  const int n0 = blockIdx.x * 128;

  v8f acc[4][2] = {};  // [m-subtile][n-subtile]

  auto load_tiles = [&](int k0, int p) {
    // A tile: 128 x 32 halves = 512 x 16B chunks, 2 per thread (async-capable)
#pragma unroll
    for (int c = 0; c < 2; ++c) {
      int chunk = tid + c * 256;        // 0..511
      int r  = chunk >> 2;              // 4 chunks per row
      int kc = (chunk & 3) * 8;
      copy_b128(&A[(size_t)(m0 + r) * K + k0 + kc], &Alds[p][r][kc]);
    }
    // B tile: 32(k) x 128(n); vector-load 8 n-values, transpose-store [n][k]
#pragma unroll
    for (int c = 0; c < 2; ++c) {
      int chunk = tid + c * 256;        // 0..511
      int kk  = chunk >> 4;             // 0..31
      int nn8 = (chunk & 15) * 8;       // 0..120
      v8h w = *(const v8h*)&W[(size_t)(k0 + kk) * N + n0 + nn8];
#pragma unroll
      for (int e = 0; e < 8; ++e) Blds[p][nn8 + e][kk] = w[e];
    }
  };

  load_tiles(0, 0);
  int p = 0;
  for (int k0 = 0; k0 < K; k0 += 32) {
    wait_async();          // this wave's async copies into buf p complete
    __syncthreads();       // everyone's tile p published; tile p^1 free
    if (k0 + 32 < K) load_tiles(k0 + 32, p ^ 1);

    v16h a0 = load_frag(&Alds[p][wm * 64 + ln][0], g);
    v16h a1 = load_frag(&Alds[p][wm * 64 + 16 + ln][0], g);
    v16h a2 = load_frag(&Alds[p][wm * 64 + 32 + ln][0], g);
    v16h a3 = load_frag(&Alds[p][wm * 64 + 48 + ln][0], g);
    v16h b0 = load_frag(&Blds[p][wn * 32 + ln][0], g);
    v16h b1 = load_frag(&Blds[p][wn * 32 + 16 + ln][0], g);
    acc[0][0] = WMMA_F32_F16(a0, b0, acc[0][0]);
    acc[0][1] = WMMA_F32_F16(a0, b1, acc[0][1]);
    acc[1][0] = WMMA_F32_F16(a1, b0, acc[1][0]);
    acc[1][1] = WMMA_F32_F16(a1, b1, acc[1][1]);
    acc[2][0] = WMMA_F32_F16(a2, b0, acc[2][0]);
    acc[2][1] = WMMA_F32_F16(a2, b1, acc[2][1]);
    acc[3][0] = WMMA_F32_F16(a3, b0, acc[3][0]);
    acc[3][1] = WMMA_F32_F16(a3, b1, acc[3][1]);
    p ^= 1;
  }

#pragma unroll
  for (int i = 0; i < 4; ++i) {
#pragma unroll
    for (int j = 0; j < 2; ++j) {
      const int col = n0 + wn * 32 + j * 16 + ln;
      const float bv = bias[col];
      if constexpr (TO_QKV) {
        _Float16* qkv = (_Float16*)outp;
        const int sel = col >> 10;  // 0=q 1=k 2=v
        const int cc  = col & (CD - 1);
        const int h   = cc >> 6, d = cc & 63;
        _Float16* dst = qkv + (size_t)sel * QKVSZ;
#pragma unroll
        for (int r = 0; r < 8; ++r) {
          int row = m0 + wm * 64 + i * 16 + 8 * g + r;  // in [0, B*T)
          int b = row >> 11, t = row & (T - 1);
          dst[((size_t)((b * H + h) * T + t)) * D + d] =
              (_Float16)(acc[i][j][r] + bv);
        }
      } else {
        float* o = (float*)outp;
#pragma unroll
        for (int r = 0; r < 8; ++r) {
          int row = m0 + wm * 64 + i * 16 + 8 * g + r;
          o[(size_t)row * N + col] = acc[i][j][r] + bv;
        }
      }
    }
  }
}

// ---------------------------------------------------------------------------
// Causal flash attention. Grid (T/64, B*H), block 128 (4 waves).
// Each wave owns 16 query rows; 32 keys per iteration; online softmax.
// ---------------------------------------------------------------------------
__global__ __launch_bounds__(128) void attn_fwd_kernel(
    const _Float16* __restrict__ q, const _Float16* __restrict__ k,
    const _Float16* __restrict__ v, _Float16* __restrict__ attn) {
  __shared__ __align__(16) _Float16 Klds[32][72];     // [key][d], 144B stride
  __shared__ __align__(16) _Float16 Vlds[64][40];     // transposed [d][key]
  __shared__ __align__(16) _Float16 Plds[4][16][40];  // per-wave P staging

  const int tid = threadIdx.x;
  const int lane = tid & 31;
  const int ln = lane & 15, g = lane >> 4;
  const int wv = tid >> 5;  // 0..3

  const int qt0 = blockIdx.x * 64;
  const int bh  = blockIdx.y;
  const int b = bh >> 4, h = bh & 15;

  const _Float16* qp = q + (size_t)bh * T * D;
  const _Float16* kp = k + (size_t)bh * T * D;
  const _Float16* vp = v + (size_t)bh * T * D;

  const int qrow0 = qt0 + wv * 16;

  // Q fragments live in registers for the whole kernel (16 rows x 64 d)
  const _Float16* qrow = qp + (size_t)(qrow0 + ln) * D;
  v16h qa0 = load_frag(qrow, g);
  v16h qa1 = load_frag(qrow + 32, g);

  v8f acc[4] = {};
  float mrow[8], lrow[8];
#pragma unroll
  for (int r = 0; r < 8; ++r) { mrow[r] = -1e30f; lrow[r] = 0.f; }

  const float sc = 0.125f;  // 1/sqrt(64)

  for (int kt0 = 0; kt0 < qt0 + 64; kt0 += 32) {
    __syncthreads();  // previous iteration's reads of K/V tiles done
    // K tile [key][d]: pure copy, async-capable, 16B chunks (2 per thread)
    for (int c = tid; c < 256; c += 128) {
      int e = c * 8, key = e >> 6, d = e & 63;
      copy_b128(&kp[(size_t)(kt0 + key) * D + d], &Klds[key][d]);
    }
    // V tile transposed to [d][key]: vector global load, scalar LDS scatter
    for (int c = tid; c < 256; c += 128) {
      int e = c * 8, key = e >> 6, d = e & 63;
      v8h val = *(const v8h*)&vp[(size_t)(kt0 + key) * D + d];
#pragma unroll
      for (int t = 0; t < 8; ++t) Vlds[d + t][key] = val[t];
    }
    wait_async();
    __syncthreads();

    // S = Q K^T (two 16x16 tiles: keys 0..15, 16..31), K-dim = D = 64
    v8f S0 = {}, S1 = {};
    {
      v16h kb = load_frag(&Klds[ln][0], g);
      S0 = WMMA_F32_F16(qa0, kb, S0);
      kb = load_frag(&Klds[ln][32], g);
      S0 = WMMA_F32_F16(qa1, kb, S0);
      kb = load_frag(&Klds[16 + ln][0], g);
      S1 = WMMA_F32_F16(qa0, kb, S1);
      kb = load_frag(&Klds[16 + ln][32], g);
      S1 = WMMA_F32_F16(qa1, kb, S1);
    }

    // online softmax in C-layout: element (m = 8g+r, n = ln)
#pragma unroll
    for (int r = 0; r < 8; ++r) {
      const int mg = qrow0 + 8 * g + r;  // global query row
      float s0 = S0[r] * sc;
      float s1 = S1[r] * sc;
      if (kt0 + ln > mg)      s0 = -1e30f;  // causal mask
      if (kt0 + 16 + ln > mg) s1 = -1e30f;
      float tmax = fmaxf(s0, s1);
#pragma unroll
      for (int w = 1; w < 16; w <<= 1)
        tmax = fmaxf(tmax, __shfl_xor(tmax, w, 32));
      const float mnew  = fmaxf(mrow[r], tmax);
      const float scale = __expf(mrow[r] - mnew);
      const float p0 = __expf(s0 - mnew);
      const float p1 = __expf(s1 - mnew);
      float ps = p0 + p1;
#pragma unroll
      for (int w = 1; w < 16; w <<= 1)
        ps += __shfl_xor(ps, w, 32);
      lrow[r] = lrow[r] * scale + ps;
      mrow[r] = mnew;
#pragma unroll
      for (int j = 0; j < 4; ++j) acc[j][r] *= scale;
      // C-layout -> A-layout via per-wave LDS (same-wave DS ops are in-order)
      Plds[wv][8 * g + r][ln]      = (_Float16)p0;
      Plds[wv][8 * g + r][16 + ln] = (_Float16)p1;
    }

    // acc += P (16x32) @ V (32x64)
    v16h pa = load_frag(&Plds[wv][ln][0], g);
#pragma unroll
    for (int j = 0; j < 4; ++j) {
      v16h vb = load_frag(&Vlds[j * 16 + ln][0], g);
      acc[j] = WMMA_F32_F16(pa, vb, acc[j]);
    }
  }

  // normalize, store f16 [B,T,C] (head-interleaved) for the proj GEMM
#pragma unroll
  for (int r = 0; r < 8; ++r) {
    const int t = qrow0 + 8 * g + r;
    const float inv = 1.0f / lrow[r];
    const size_t base = ((size_t)(b * T + t) << 10) + (h << 6);
#pragma unroll
    for (int j = 0; j < 4; ++j)
      attn[base + j * 16 + ln] = (_Float16)(acc[j][r] * inv);
  }
}

// ---------------------------------------------------------------------------
extern "C" void kernel_launch(void* const* d_in, const int* in_sizes, int n_in,
                              void* d_out, int out_size, void* d_ws, size_t ws_size,
                              hipStream_t stream) {
  (void)in_sizes; (void)n_in; (void)out_size; (void)ws_size;
  const float* x  = (const float*)d_in[0];
  const float* Wa = (const float*)d_in[1];
  const float* ba = (const float*)d_in[2];
  const float* Wp = (const float*)d_in[3];
  const float* bp = (const float*)d_in[4];
  float* out = (float*)d_out;

  // f16 workspace layout (element offsets)
  _Float16* ws = (_Float16*)d_ws;
  _Float16* xh  = ws;                         // 4096x1024
  _Float16* Wah = xh  + (long)MROWS * CD;     // 1024x3072
  _Float16* Wph = Wah + (long)CD * 3 * CD;    // 1024x1024
  _Float16* qw  = Wph + (long)CD * CD;        // [B,H,T,D]
  _Float16* kw  = qw + QKVSZ;
  _Float16* vw  = kw + QKVSZ;
  _Float16* aw  = vw + QKVSZ;                 // [B,T,C]

  // 0) one-time f32 -> f16 conversions (bandwidth-bound pre-pass)
  cvt_f16_kernel<<<(MROWS * CD / 4) / 256, 256, 0, stream>>>(x, xh, MROWS * CD / 4);
  cvt_f16_kernel<<<(CD * 3 * CD / 4) / 256, 256, 0, stream>>>(Wa, Wah, CD * 3 * CD / 4);
  cvt_f16_kernel<<<(CD * CD / 4) / 256, 256, 0, stream>>>(Wp, Wph, CD * CD / 4);

  // 1) QKV projection: [4096,1024] x [1024,3072]
  gemm_wmma_kernel<1024, 3072, true>
      <<<dim3(3072 / 128, MROWS / 128), 256, 0, stream>>>(xh, Wah, ba, (void*)qw);

  // 2) causal flash attention per (b,h)
  attn_fwd_kernel<<<dim3(T / 64, BSZ * H), 128, 0, stream>>>(qw, kw, vw, aw);

  // 3) output projection: [4096,1024] x [1024,1024]
  gemm_wmma_kernel<1024, 1024, false>
      <<<dim3(1024 / 128, MROWS / 128), 256, 0, stream>>>(aw, Wph, bp, (void*)out);
}
